// GATBlock_2559800509145
// MI455X (gfx1250) — compile-verified, matrix operand order checked by reading, not run on hardware
//
#include <hip/hip_runtime.h>
#include <hip/hip_bf16.h>

typedef __attribute__((ext_vector_type(16))) __bf16 v16bf;
typedef __attribute__((ext_vector_type(8)))  __bf16 v8bf;
typedef __attribute__((ext_vector_type(8)))  float  v8f;

#define NN   10000
#define NE   160000
#define DD   768
#define H1n  4
#define C1n  192
#define H2n  2
#define C2n  96
#define HC1  (H1n*C1n)   // 768
#define HC2  (H2n*C2n)   // 192
#define DK   (DD+HC2)    // 960

// ---------------------------------------------------------------------------
// ordered-uint mapping for float atomic max
// ---------------------------------------------------------------------------
__device__ __forceinline__ unsigned ford(float f) {
    unsigned u = __float_as_uint(f);
    return (u & 0x80000000u) ? ~u : (u | 0x80000000u);
}
__device__ __forceinline__ float funord(unsigned u) {
    return (u & 0x80000000u) ? __uint_as_float(u ^ 0x80000000u)
                             : __uint_as_float(~u);
}

// ---------------------------------------------------------------------------
// async global -> LDS (16B per lane), tracked by ASYNCcnt
// ---------------------------------------------------------------------------
__device__ __forceinline__ void async_b128_to_lds(const void* gptr, void* lptr) {
    unsigned lds_off = (unsigned)(size_t)lptr;          // low 32b = LDS byte offset
    unsigned long long ga = (unsigned long long)gptr;
    asm volatile("global_load_async_to_lds_b128 %0, %1, off"
                 :: "v"(lds_off), "v"(ga) : "memory");
}
__device__ __forceinline__ void wait_asynccnt0() {
    asm volatile("s_wait_asynccnt 0x0" ::: "memory");
}

// ---------------------------------------------------------------------------
// bf16 WMMA GEMM:  C[M,Nc] = A[M,K] * Bt[Nc,K]^T  (+bias), f32 accum
// workgroup (8 waves) owns 256 rows x 64 cols; B strip staged in LDS via
// double-buffered async copies; each wave computes a 32x64 tile:
// 2 A fragments x 4 LDS B fragments -> 8 v_wmma per K-step.
// ---------------------------------------------------------------------------
__global__ __launch_bounds__(256)
void gat_gemm_bf16(const __bf16* __restrict__ A, const __bf16* __restrict__ Bt,
                   const float* __restrict__ bias, float* __restrict__ C,
                   int M, int Nc, int K)
{
    __shared__ __bf16 Bs[2][64][32];                    // 2 x 4KB double buffer

    const int nGroups = Nc >> 6;
    const int mSuper  = blockIdx.x / nGroups;
    const int nG      = blockIdx.x % nGroups;
    const int wave    = threadIdx.x >> 5;
    const int lane    = threadIdx.x & 31;
    const int hf      = lane >> 4;                      // K-half per ISA A layout
    const int l16     = lane & 15;

    const int mBase = mSuper * 256 + wave * 32;
    const int nBase = nG << 6;

    int mA0 = mBase + l16;        if (mA0 > M - 1) mA0 = M - 1;
    int mA1 = mBase + 16 + l16;   if (mA1 > M - 1) mA1 = M - 1;
    const __bf16* __restrict__ aRow0 = A + (size_t)mA0 * K;
    const __bf16* __restrict__ aRow1 = A + (size_t)mA1 * K;

    // staging assignment: thread -> (row, 16B chunk) of the 64x32 B tile
    const int sr = threadIdx.x >> 2;                    // 0..63
    const int sc = (threadIdx.x & 3) * 8;               // element offset 0/8/16/24

    // prologue: stage k=0 into buffer 0
    async_b128_to_lds(Bt + (size_t)(nBase + sr) * K + sc, &Bs[0][sr][sc]);

    v8f acc0[4] = {};
    v8f acc1[4] = {};

    int buf = 0;
    for (int k = 0; k < K; k += 32) {
        wait_asynccnt0();                               // staging of `buf` done
        __syncthreads();                                // visible to all waves

        if (k + 32 < K) {                               // stage next K into other buffer
            async_b128_to_lds(Bt + (size_t)(nBase + sr) * K + (k + 32) + sc,
                              &Bs[buf ^ 1][sr][sc]);
        }

        __builtin_prefetch(aRow0 + k + 64, 0, 0);
        __builtin_prefetch(aRow1 + k + 64, 0, 0);

        v8bf alo0 = *(const v8bf*)(aRow0 + k + hf * 8);
        v8bf ahi0 = *(const v8bf*)(aRow0 + k + 16 + hf * 8);
        v16bf af0 = __builtin_shufflevector(alo0, ahi0,
            0,1,2,3,4,5,6,7,8,9,10,11,12,13,14,15);
        v8bf alo1 = *(const v8bf*)(aRow1 + k + hf * 8);
        v8bf ahi1 = *(const v8bf*)(aRow1 + k + 16 + hf * 8);
        v16bf af1 = __builtin_shufflevector(alo1, ahi1,
            0,1,2,3,4,5,6,7,8,9,10,11,12,13,14,15);

        #pragma unroll
        for (int t = 0; t < 4; ++t) {
            v8bf blo = *(const v8bf*)&Bs[buf][t * 16 + l16][hf * 8];       // ds_load_b128
            v8bf bhi = *(const v8bf*)&Bs[buf][t * 16 + l16][16 + hf * 8];  // ds_load_b128
            v16bf bf = __builtin_shufflevector(blo, bhi,
                0,1,2,3,4,5,6,7,8,9,10,11,12,13,14,15);
            acc0[t] = __builtin_amdgcn_wmma_f32_16x16x32_bf16(
                false, af0, false, bf, (short)0, acc0[t], false, false);
            acc1[t] = __builtin_amdgcn_wmma_f32_16x16x32_bf16(
                false, af1, false, bf, (short)0, acc1[t], false, false);
        }

        __syncthreads();                                // all waves done reading buf
        buf ^= 1;
    }

    // C/D layout: VGPR v -> m = mBase(+16) + v + 8*hf ; n = nBase + t*16 + l16
    #pragma unroll
    for (int t = 0; t < 4; ++t) {
        const int n = nBase + t * 16 + l16;
        const float bv = bias ? bias[n] : 0.0f;
        #pragma unroll
        for (int v = 0; v < 8; ++v) {
            const int m0 = mBase + v + 8 * hf;
            const int m1 = mBase + 16 + v + 8 * hf;
            if (m0 < M) C[(size_t)m0 * Nc + n] = acc0[t][v] + bv;
            if (m1 < M) C[(size_t)m1 * Nc + n] = acc1[t][v] + bv;
        }
    }
}

// ---------------------------------------------------------------------------
// LayerNorm(x) -> bf16
// ---------------------------------------------------------------------------
__global__ __launch_bounds__(256)
void gat_ln_bf16(const float* __restrict__ x, const float* __restrict__ g,
                 const float* __restrict__ b, __bf16* __restrict__ y, int D)
{
    __shared__ float r0[256];
    __shared__ float r1[256];
    const int n = blockIdx.x;
    const float* row = x + (size_t)n * D;
    float s = 0.f, ss = 0.f;
    for (int i = threadIdx.x; i < D; i += 256) { float v = row[i]; s += v; ss += v * v; }
    r0[threadIdx.x] = s; r1[threadIdx.x] = ss;
    __syncthreads();
    for (int o = 128; o > 0; o >>= 1) {
        if ((int)threadIdx.x < o) { r0[threadIdx.x] += r0[threadIdx.x + o];
                                    r1[threadIdx.x] += r1[threadIdx.x + o]; }
        __syncthreads();
    }
    const float mu  = r0[0] / D;
    const float var = r1[0] / D - mu * mu;
    const float inv = rsqrtf(var + 1e-5f);
    for (int i = threadIdx.x; i < D; i += 256)
        y[(size_t)n * D + i] = (__bf16)((row[i] - mu) * inv * g[i] + b[i]);
}

// ---------------------------------------------------------------------------
// per-(node,head) attention logits via wave reduction
// ---------------------------------------------------------------------------
__global__ __launch_bounds__(256)
void gat_attn_logits(const float* __restrict__ h, const float* __restrict__ aS,
                     const float* __restrict__ aD, float* __restrict__ sS,
                     float* __restrict__ sD, int N, int H, int C)
{
    const int wave = (blockIdx.x * blockDim.x + threadIdx.x) >> 5;
    const int lane = threadIdx.x & 31;
    if (wave >= N * H) return;
    const int node = wave / H, head = wave % H;
    const float* hp = h + (size_t)node * H * C + head * C;
    float vs = 0.f, vd = 0.f;
    for (int c = lane; c < C; c += 32) {
        const float hv = hp[c];
        vs += hv * aS[head * C + c];
        vd += hv * aD[head * C + c];
    }
    for (int o = 16; o; o >>= 1) { vs += __shfl_xor(vs, o, 32);
                                   vd += __shfl_xor(vd, o, 32); }
    if (lane == 0) { sS[wave] = vs; sD[wave] = vd; }
}

// ---------------------------------------------------------------------------
// edge pass 1: e = leaky_relu(sS[src]+sD[dst]); segment max via ordered atomic
// ---------------------------------------------------------------------------
__global__ __launch_bounds__(256)
void gat_edge_scores(const long long* __restrict__ ei, int E, int N, int H,
                     const float* __restrict__ sS, const float* __restrict__ sD,
                     float* __restrict__ e, unsigned* __restrict__ emax)
{
    const int tid = blockIdx.x * 256 + threadIdx.x;
    const int tot = (E + N) * H;
    if (tid >= tot) return;
    const int edge = tid / H, hh = tid % H;
    long long s, d;
    if (edge < E) { s = ei[edge]; d = ei[E + edge]; } else { s = d = edge - E; }
    float v = sS[(size_t)s * H + hh] + sD[(size_t)d * H + hh];
    v = (v < 0.f) ? 0.2f * v : v;
    e[tid] = v;
    atomicMax(&emax[(size_t)d * H + hh], ford(v));
}

// ---------------------------------------------------------------------------
// edge pass 2: ee = exp(e - max); segment sum via atomic add
// ---------------------------------------------------------------------------
__global__ __launch_bounds__(256)
void gat_edge_exp(const long long* __restrict__ ei, int E, int N, int H,
                  float* __restrict__ e, const unsigned* __restrict__ emax,
                  float* __restrict__ den)
{
    const int tid = blockIdx.x * 256 + threadIdx.x;
    const int tot = (E + N) * H;
    if (tid >= tot) return;
    const int edge = tid / H, hh = tid % H;
    long long d;
    if (edge < E) { d = ei[E + edge]; } else { d = edge - E; }
    const float m = funord(emax[(size_t)d * H + hh]);
    const float ee = __expf(e[tid] - m);
    e[tid] = ee;
    atomicAdd(&den[(size_t)d * H + hh], ee);
}

// ---------------------------------------------------------------------------
// edge pass 3: out[dst] += alpha * h[src], 4 channels / thread
// ---------------------------------------------------------------------------
__global__ __launch_bounds__(256)
void gat_edge_scatter(const long long* __restrict__ ei, int E, int N, int H, int C,
                      const float* __restrict__ ee, const float* __restrict__ den,
                      const float* __restrict__ hf, float* __restrict__ outp)
{
    const int HC4 = (H * C) >> 2;
    const int gid = blockIdx.x * 256 + threadIdx.x;
    const int tot = (E + N) * HC4;
    if (gid >= tot) return;
    const int edge = gid / HC4;
    const int j = (gid % HC4) << 2;
    const int head = j / C;
    long long s, d;
    if (edge < E) { s = ei[edge]; d = ei[E + edge]; } else { s = d = edge - E; }
    const float alpha = ee[(size_t)edge * H + head] / den[(size_t)d * H + head];
    const float4 hv = *(const float4*)(hf + (size_t)s * H * C + j);
    float* op = outp + (size_t)d * H * C + j;
    atomicAdd(op + 0, alpha * hv.x);
    atomicAdd(op + 1, alpha * hv.y);
    atomicAdd(op + 2, alpha * hv.z);
    atomicAdd(op + 3, alpha * hv.w);
}

// ---------------------------------------------------------------------------
// small utility kernels
// ---------------------------------------------------------------------------
__global__ void gat_fill32(unsigned* p, unsigned v, int n) {
    const int i = blockIdx.x * 256 + threadIdx.x;
    if (i < n) p[i] = v;
}
__global__ void gat_init_rows(float* __restrict__ o, const float* __restrict__ bias,
                              int N, int HC) {
    const int i = blockIdx.x * 256 + threadIdx.x;
    if (i < N * HC) o[i] = bias[i % HC];
}
__global__ void gat_transpose_bf16(const float* __restrict__ W, __bf16* __restrict__ Wt,
                                   int K, int Nc) {
    const int i = blockIdx.x * 256 + threadIdx.x;
    if (i < K * Nc) {
        const int k = i / Nc, n = i % Nc;
        Wt[(size_t)n * K + k] = (__bf16)W[i];
    }
}
__global__ void gat_to_bf16(const float* __restrict__ a, __bf16* __restrict__ b, int n) {
    const int i = blockIdx.x * 256 + threadIdx.x;
    if (i < n) b[i] = (__bf16)a[i];
}
__global__ void gat_concat_bf16(const float* __restrict__ x, const float* __restrict__ o2,
                                __bf16* __restrict__ comb, int N, int D, int D2) {
    const int i = blockIdx.x * 256 + threadIdx.x;
    const int DK_ = D + D2;
    if (i < N * DK_) {
        const int n = i / DK_, c = i % DK_;
        comb[i] = (__bf16)((c < D) ? x[(size_t)n * D + c]
                                   : o2[(size_t)n * D2 + (c - D)]);
    }
}

// ---------------------------------------------------------------------------
// host driver
// ---------------------------------------------------------------------------
static inline int cdiv(long long a, long long b) { return (int)((a + b - 1) / b); }

extern "C" void kernel_launch(void* const* d_in, const int* in_sizes, int n_in,
                              void* d_out, int out_size, void* d_ws, size_t ws_size,
                              hipStream_t stream)
{
    const float*     x   = (const float*)d_in[0];
    const long long* ei  = (const long long*)d_in[1];
    const float*     ng  = (const float*)d_in[2];
    const float*     nb  = (const float*)d_in[3];
    const float*     W1  = (const float*)d_in[4];
    const float*     as1 = (const float*)d_in[5];
    const float*     ad1 = (const float*)d_in[6];
    const float*     b1  = (const float*)d_in[7];
    const float*     W2  = (const float*)d_in[8];
    const float*     as2 = (const float*)d_in[9];
    const float*     ad2 = (const float*)d_in[10];
    const float*     b2  = (const float*)d_in[11];
    const float*     Wu  = (const float*)d_in[12];
    const float*     bu  = (const float*)d_in[13];
    float* out = (float*)d_out;

    const int Etot = NE + NN;

    // bump allocator over workspace (256B aligned)
    char* wsb = (char*)d_ws;
    size_t off = 0;
    auto alloc = [&](size_t bytes) -> char* {
        char* p = wsb + off;
        off += (bytes + 255) & ~(size_t)255;
        return p;
    };

    __bf16* xnbf = (__bf16*)alloc((size_t)NN * DD * 2);   // reused as o1bf
    __bf16* w1t  = (__bf16*)alloc((size_t)DD * HC1 * 2);
    float*  h1   = (float*) alloc((size_t)NN * HC1 * 4);  // reused as comb
    float*  ss1  = (float*) alloc((size_t)NN * H1n * 4);
    float*  sd1  = (float*) alloc((size_t)NN * H1n * 4);
    float*  e1   = (float*) alloc((size_t)Etot * H1n * 4);
    unsigned* emax1 = (unsigned*)alloc((size_t)NN * H1n * 4);
    float*  den1 = (float*) alloc((size_t)NN * H1n * 4);
    float*  out1 = (float*) alloc((size_t)NN * HC1 * 4);  // reused as out2
    __bf16* w2t  = (__bf16*)alloc((size_t)DD * HC2 * 2);
    float*  h2   = (float*) alloc((size_t)NN * HC2 * 4);
    float*  ss2  = (float*) alloc((size_t)NN * H2n * 4);
    float*  sd2  = (float*) alloc((size_t)NN * H2n * 4);
    float*  e2   = (float*) alloc((size_t)Etot * H2n * 4);
    unsigned* emax2 = (unsigned*)alloc((size_t)NN * H2n * 4);
    float*  den2 = (float*) alloc((size_t)NN * H2n * 4);
    __bf16* wut  = (__bf16*)alloc((size_t)DK * DD * 2);

    __bf16* o1bf = xnbf;           // x_norm_bf dead after GEMM1
    __bf16* comb = (__bf16*)h1;    // h1 dead after scatter1
    float*  out2 = out1;           // out1 dead after conversion to o1bf

    // --- weight prep ---
    gat_transpose_bf16<<<cdiv((long long)DD * HC1, 256), 256, 0, stream>>>(W1, w1t, DD, HC1);
    gat_transpose_bf16<<<cdiv((long long)DD * HC2, 256), 256, 0, stream>>>(W2, w2t, DD, HC2);
    gat_transpose_bf16<<<cdiv((long long)DK * DD, 256), 256, 0, stream>>>(Wu, wut, DK, DD);

    // --- LayerNorm -> bf16 ---
    gat_ln_bf16<<<NN, 256, 0, stream>>>(x, ng, nb, xnbf, DD);

    // --- GEMM1: h1 = x_norm @ W1 ---
    gat_gemm_bf16<<<cdiv(NN, 256) * (HC1 / 64), 256, 0, stream>>>(
        xnbf, w1t, nullptr, h1, NN, HC1, DD);

    // --- GAT layer 1 edge phase ---
    gat_attn_logits<<<cdiv((long long)NN * H1n * 32, 256), 256, 0, stream>>>(
        h1, as1, ad1, ss1, sd1, NN, H1n, C1n);
    gat_fill32<<<cdiv(NN * H1n, 256), 256, 0, stream>>>(emax1, 0u, NN * H1n);
    gat_fill32<<<cdiv(NN * H1n, 256), 256, 0, stream>>>((unsigned*)den1, 0u, NN * H1n);
    gat_init_rows<<<cdiv((long long)NN * HC1, 256), 256, 0, stream>>>(out1, b1, NN, HC1);
    gat_edge_scores<<<cdiv((long long)Etot * H1n, 256), 256, 0, stream>>>(
        ei, NE, NN, H1n, ss1, sd1, e1, emax1);
    gat_edge_exp<<<cdiv((long long)Etot * H1n, 256), 256, 0, stream>>>(
        ei, NE, NN, H1n, e1, emax1, den1);
    gat_edge_scatter<<<cdiv((long long)Etot * (HC1 / 4), 256), 256, 0, stream>>>(
        ei, NE, NN, H1n, C1n, e1, den1, h1, out1);

    // --- GEMM2: h2 = out1 @ W2 ---
    gat_to_bf16<<<cdiv((long long)NN * HC1, 256), 256, 0, stream>>>(out1, o1bf, NN * HC1);
    gat_gemm_bf16<<<cdiv(NN, 256) * (HC2 / 64), 256, 0, stream>>>(
        o1bf, w2t, nullptr, h2, NN, HC2, DD);

    // --- GAT layer 2 edge phase ---
    gat_attn_logits<<<cdiv((long long)NN * H2n * 32, 256), 256, 0, stream>>>(
        h2, as2, ad2, ss2, sd2, NN, H2n, C2n);
    gat_fill32<<<cdiv(NN * H2n, 256), 256, 0, stream>>>(emax2, 0u, NN * H2n);
    gat_fill32<<<cdiv(NN * H2n, 256), 256, 0, stream>>>((unsigned*)den2, 0u, NN * H2n);
    gat_init_rows<<<cdiv((long long)NN * HC2, 256), 256, 0, stream>>>(out2, b2, NN, HC2);
    gat_edge_scores<<<cdiv((long long)Etot * H2n, 256), 256, 0, stream>>>(
        ei, NE, NN, H2n, ss2, sd2, e2, emax2);
    gat_edge_exp<<<cdiv((long long)Etot * H2n, 256), 256, 0, stream>>>(
        ei, NE, NN, H2n, e2, emax2, den2);
    gat_edge_scatter<<<cdiv((long long)Etot * (HC2 / 4), 256), 256, 0, stream>>>(
        ei, NE, NN, H2n, C2n, e2, den2, h2, out2);

    // --- final projection: out = [x ‖ out2] @ Wu + bu ---
    gat_concat_bf16<<<cdiv((long long)NN * DK, 256), 256, 0, stream>>>(
        x, out2, comb, NN, DD, HC2);
    gat_gemm_bf16<<<cdiv(NN, 256) * (DD / 64), 256, 0, stream>>>(
        comb, wut, bu, out, NN, DD, DK);

    (void)in_sizes; (void)n_in; (void)out_size; (void)ws_size;
}
